// MultiStageDiT_4964982194894
// MI455X (gfx1250) — matrix operand order repaired and checked by reading, not compiled, not run on hardware
//
#include <hip/hip_runtime.h>
#include <hip/hip_bf16.h>

// MI455X / gfx1250 DiT block: all GEMMs on v_wmma_f32_16x16x32_bf16 (wave32),
// double-buffered global->LDS staging via gfx1250 async-to-LDS ops (ASYNCcnt).

typedef __attribute__((ext_vector_type(16))) __bf16 v16bf;
typedef __attribute__((ext_vector_type(8)))  float  v8f;

#define D_MODEL 1024
#define T_TOK   4096
#define N_BATCH 4
#define M_ROWS  (N_BATCH * T_TOK)     // 16384
#define WS      16
#define NH      16
#define HD      64
#define L_WIN   256
#define NWIN    64

static __device__ __forceinline__ int win_row(int n, int t) {
    int i = t >> 6, j = t & 63;
    int win = ((n * 4 + (i >> 4)) * 4) + (j >> 4);
    int l   = (i & 15) * 16 + (j & 15);
    return win * L_WIN + l;
}

// gfx1250 async global->LDS copy (16B per lane), tracked by ASYNCcnt.
static __device__ __forceinline__ void async_copy_b128(__bf16* lds_dst,
                                                       const __bf16* gsrc) {
    unsigned lds_off = (unsigned)(size_t)lds_dst;  // low 32 bits = LDS offset
    asm volatile("global_load_async_to_lds_b128 %0, %1, off"
                 :: "v"(lds_off), "v"(gsrc) : "memory");
}
static __device__ __forceinline__ void wait_async0() {
    asm volatile("s_wait_asynccnt 0x0" ::: "memory");
}

// ---------------------------------------------------------------- f32 -> bf16
__global__ __launch_bounds__(256) void f2bf_kernel(const float* __restrict__ in,
                                                   __bf16* __restrict__ out, int n) {
    int i = blockIdx.x * 256 + threadIdx.x;
    if (i < n) out[i] = (__bf16)in[i];
}

// ------------------------------------------------- mod = silu(c) @ W^T + b
__global__ __launch_bounds__(256) void mod_kernel(const float* __restrict__ c,
                                                  const float* __restrict__ w,
                                                  const float* __restrict__ b,
                                                  float* __restrict__ mod) {
    __shared__ float sc[D_MODEL];
    int n = blockIdx.y;
    for (int i = threadIdx.x; i < D_MODEL; i += 256) {
        float v = c[n * D_MODEL + i];
        sc[i] = v / (1.0f + __expf(-v));
    }
    __syncthreads();
    int f = blockIdx.x * 256 + threadIdx.x;
    const float* wr = w + (size_t)f * D_MODEL;
    float s = b[f];
    for (int k = 0; k < D_MODEL; ++k) s += sc[k] * wr[k];
    mod[n * 6 * D_MODEL + f] = s;
}

// -------------------------------- LN + modulate -> bf16 (optional window order)
__global__ __launch_bounds__(256) void ln_mod_kernel(const float* __restrict__ x,
                                                     const float* __restrict__ mod,
                                                     int shOff, int scOff,
                                                     __bf16* __restrict__ out,
                                                     int windowOrder) {
    int row = blockIdx.x;
    int n = row >> 12, t = row & 4095;
    const float* xr = x + (size_t)row * D_MODEL;

    float xv[4];
    float lsum = 0.f, lsq = 0.f;
    #pragma unroll
    for (int u = 0; u < 4; ++u) {
        float v = xr[threadIdx.x + u * 256];
        xv[u] = v; lsum += v; lsq += v * v;
    }
    #pragma unroll
    for (int off = 16; off > 0; off >>= 1) {
        lsum += __shfl_xor(lsum, off, 32);
        lsq  += __shfl_xor(lsq,  off, 32);
    }
    __shared__ float rs[8], rq[8];
    if ((threadIdx.x & 31) == 0) { rs[threadIdx.x >> 5] = lsum; rq[threadIdx.x >> 5] = lsq; }
    __syncthreads();
    float sum = 0.f, sq = 0.f;
    #pragma unroll
    for (int wv = 0; wv < 8; ++wv) { sum += rs[wv]; sq += rq[wv]; }
    float mean = sum * (1.0f / D_MODEL);
    float var  = sq  * (1.0f / D_MODEL) - mean * mean;
    float rstd = rsqrtf(var + 1e-6f);

    size_t orow = windowOrder ? (size_t)win_row(n, t) : (size_t)row;
    const float* shp = mod + (size_t)n * 6 * D_MODEL + shOff;
    const float* scp = mod + (size_t)n * 6 * D_MODEL + scOff;
    #pragma unroll
    for (int u = 0; u < 4; ++u) {
        int i = threadIdx.x + u * 256;
        float v = (xv[u] - mean) * rstd;
        v = v * (1.0f + scp[i]) + shp[i];
        out[orow * D_MODEL + i] = (__bf16)v;
    }
}

// --------------------------------------------- bf16 WMMA GEMM: C = A @ W^T
// A:[M,K], W:[N,K] bf16 row-major. Block tile 128x128, 8 waves.
// Double-buffered async global->LDS pipeline: copy chunk k+1 overlaps WMMA on k.
template <int ACT, bool HASB, bool OF32, bool OBF>
__global__ __launch_bounds__(256) void gemm_bf16_kernel(
        const __bf16* __restrict__ A, const __bf16* __restrict__ W,
        const float* __restrict__ bias, float* __restrict__ Cf,
        __bf16* __restrict__ Cb, int M, int N, int K) {
    __shared__ __attribute__((aligned(16))) __bf16 As[2][128][40];   // +8 pad
    __shared__ __attribute__((aligned(16))) __bf16 Wsh[2][128][40];
    int tid  = threadIdx.x;
    int lane = tid & 31, wave = tid >> 5;
    int laneR = lane & 15, laneG = lane >> 4;
    int m0 = blockIdx.y * 128, n0 = blockIdx.x * 128;
    int r = tid >> 1, cc = (tid & 1) * 16;

    const __bf16* ga = A + (size_t)(m0 + r) * K + cc;
    const __bf16* gw = W + (size_t)(n0 + r) * K + cc;

    // prologue: stage chunk 0 into buffer 0
    async_copy_b128(&As[0][r][cc],      ga);
    async_copy_b128(&As[0][r][cc + 8],  ga + 8);
    async_copy_b128(&Wsh[0][r][cc],     gw);
    async_copy_b128(&Wsh[0][r][cc + 8], gw + 8);

    v8f acc[8] = {};
    int buf = 0;
    for (int kc = 0; kc < K; kc += 32) {
        wait_async0();          // this wave's copies into `buf` are done
        __syncthreads();        // publish LDS to all waves; prior reads of buf^1 done
        if (kc + 32 < K) {      // prefetch next chunk into the other buffer
            int nk = kc + 32;
            async_copy_b128(&As[buf ^ 1][r][cc],      ga + nk);
            async_copy_b128(&As[buf ^ 1][r][cc + 8],  ga + nk + 8);
            async_copy_b128(&Wsh[buf ^ 1][r][cc],     gw + nk);
            async_copy_b128(&Wsh[buf ^ 1][r][cc + 8], gw + nk + 8);
        }
        v16bf af;
        int ar = wave * 16 + laneR;
        #pragma unroll
        for (int e = 0; e < 16; ++e)
            af[e] = As[buf][ar][e + laneG * 16];
        #pragma unroll
        for (int ct = 0; ct < 8; ++ct) {
            v16bf bfr;
            int br = ct * 16 + laneR;
            #pragma unroll
            for (int e = 0; e < 16; ++e) bfr[e] = Wsh[buf][br][e + laneG * 16];
            acc[ct] = __builtin_amdgcn_wmma_f32_16x16x32_bf16(
                          false, af, false, bfr, (short)0, acc[ct], false, false);
        }
        buf ^= 1;
    }

    float bv[8];
    if (HASB) {
        #pragma unroll
        for (int ct = 0; ct < 8; ++ct) bv[ct] = bias[n0 + ct * 16 + laneR];
    }
    #pragma unroll
    for (int ct = 0; ct < 8; ++ct) {
        #pragma unroll
        for (int e = 0; e < 8; ++e) {
            int row = m0 + wave * 16 + e + laneG * 8;
            int col = n0 + ct * 16 + laneR;
            float v = acc[ct][e];
            if (HASB) v += bv[ct];
            if (ACT == 1) {  // tanh-approx GELU
                float t = tanhf(0.7978845608028654f * (v + 0.044715f * v * v * v));
                v = 0.5f * v * (1.0f + t);
            }
            size_t idx = (size_t)row * N + col;
            if (OF32) Cf[idx] = v;
            if (OBF)  Cb[idx] = (__bf16)v;
        }
    }
}

// -------------------------------------- fused per-(window,head) attention
__global__ __launch_bounds__(256) void attn_kernel(
        const float* __restrict__ qkv,   // [16384][3072] window-ordered rows
        const float* __restrict__ rph, const float* __restrict__ rpw, // [31][64]
        __bf16* __restrict__ out,        // [16384][1024] window-ordered
        float scale) {
    __shared__ __attribute__((aligned(16))) __bf16 ks[L_WIN][HD + 8];
    __shared__ __attribute__((aligned(16))) __bf16 vsT[HD][L_WIN + 8];
    __shared__ float relh[L_WIN][16];
    __shared__ float relw[L_WIN][16];
    __shared__ __bf16 pls[8][16][L_WIN + 8];

    int win = blockIdx.x >> 4, head = blockIdx.x & 15;
    const float* qbase = qkv + (size_t)win * L_WIN * (3 * D_MODEL) + head * HD;
    const float* kbase = qbase + D_MODEL;
    const float* vbase = qbase + 2 * D_MODEL;
    int tid = threadIdx.x;

    // stage K (row-major) and V (transposed) as bf16 in LDS
    for (int base = tid * 4; base < L_WIN * HD; base += 1024) {
        int row = base >> 6, d = base & 63;
        float4 k4 = *(const float4*)(kbase + (size_t)row * (3 * D_MODEL) + d);
        float4 v4 = *(const float4*)(vbase + (size_t)row * (3 * D_MODEL) + d);
        ks[row][d + 0] = (__bf16)k4.x; ks[row][d + 1] = (__bf16)k4.y;
        ks[row][d + 2] = (__bf16)k4.z; ks[row][d + 3] = (__bf16)k4.w;
        vsT[d + 0][row] = (__bf16)v4.x; vsT[d + 1][row] = (__bf16)v4.y;
        vsT[d + 2][row] = (__bf16)v4.z; vsT[d + 3][row] = (__bf16)v4.w;
    }
    {   // decomposed rel-pos bias, one query row per thread (UNscaled q)
        int l = tid, qh = l >> 4, qw = l & 15;
        float qv[HD];
        #pragma unroll
        for (int d = 0; d < HD; ++d) qv[d] = qbase[(size_t)l * (3 * D_MODEL) + d];
        for (int kh = 0; kh < 16; ++kh) {
            const float* rr = rph + (qh - kh + WS - 1) * HD;
            float s = 0.f;
            #pragma unroll
            for (int d = 0; d < HD; ++d) s += qv[d] * rr[d];
            relh[l][kh] = s;
        }
        for (int kw = 0; kw < 16; ++kw) {
            const float* rr = rpw + (qw - kw + WS - 1) * HD;
            float s = 0.f;
            #pragma unroll
            for (int d = 0; d < HD; ++d) s += qv[d] * rr[d];
            relw[l][kw] = s;
        }
    }
    __syncthreads();

    int lane = tid & 31, wave = tid >> 5;
    int laneR = lane & 15, laneG = lane >> 4;

    for (int rb = 0; rb < 2; ++rb) {
        int rowBase = rb * 128 + wave * 16;
        // ---- S = q k^T (16 col tiles)
        v8f s[16] = {};
        for (int kc = 0; kc < HD; kc += 32) {
            v16bf af;
            const float* qrow = qbase + (size_t)(rowBase + laneR) * (3 * D_MODEL) + kc;
            #pragma unroll
            for (int e = 0; e < 16; ++e)
                af[e] = (__bf16)qrow[e + laneG * 16];
            #pragma unroll
            for (int ct = 0; ct < 16; ++ct) {
                v16bf bfr;
                #pragma unroll
                for (int e = 0; e < 16; ++e)
                    bfr[e] = ks[ct * 16 + laneR][kc + e + laneG * 16];
                s[ct] = __builtin_amdgcn_wmma_f32_16x16x32_bf16(
                            false, af, false, bfr, (short)0, s[ct], false, false);
            }
        }
        // ---- scale + rel bias (col = ct*16+laneR -> kh=ct, kw=laneR)
        #pragma unroll
        for (int ct = 0; ct < 16; ++ct)
            #pragma unroll
            for (int e = 0; e < 8; ++e) {
                int row = rowBase + e + 8 * laneG;
                s[ct][e] = s[ct][e] * scale + relh[row][ct] + relw[row][laneR];
            }
        // ---- softmax (row lives in one 16-lane half across 16 tiles)
        #pragma unroll
        for (int e = 0; e < 8; ++e) {
            float m = -1e30f;
            #pragma unroll
            for (int ct = 0; ct < 16; ++ct) m = fmaxf(m, s[ct][e]);
            #pragma unroll
            for (int off = 1; off < 16; off <<= 1) m = fmaxf(m, __shfl_xor(m, off, 32));
            float sm = 0.f;
            #pragma unroll
            for (int ct = 0; ct < 16; ++ct) {
                float p = __expf(s[ct][e] - m);
                s[ct][e] = p; sm += p;
            }
            #pragma unroll
            for (int off = 1; off < 16; off <<= 1) sm += __shfl_xor(sm, off, 32);
            float inv = 1.0f / sm;
            #pragma unroll
            for (int ct = 0; ct < 16; ++ct) s[ct][e] *= inv;
        }
        // ---- stage P (C-layout -> A-layout via LDS)
        #pragma unroll
        for (int ct = 0; ct < 16; ++ct)
            #pragma unroll
            for (int e = 0; e < 8; ++e)
                pls[wave][e + 8 * laneG][ct * 16 + laneR] = (__bf16)s[ct][e];
        // ---- O = P @ V
        v8f o[4] = {};
        for (int kc = 0; kc < L_WIN; kc += 32) {
            v16bf af;
            #pragma unroll
            for (int e = 0; e < 16; ++e)
                af[e] = pls[wave][laneR][kc + e + laneG * 16];
            #pragma unroll
            for (int ct = 0; ct < 4; ++ct) {
                v16bf bfr;
                #pragma unroll
                for (int e = 0; e < 16; ++e)
                    bfr[e] = vsT[ct * 16 + laneR][kc + e + laneG * 16];
                o[ct] = __builtin_amdgcn_wmma_f32_16x16x32_bf16(
                            false, af, false, bfr, (short)0, o[ct], false, false);
            }
        }
        #pragma unroll
        for (int ct = 0; ct < 4; ++ct)
            #pragma unroll
            for (int e = 0; e < 8; ++e) {
                size_t row = (size_t)win * L_WIN + rowBase + e + 8 * laneG;
                int col = head * HD + ct * 16 + laneR;
                out[row * D_MODEL + col] = (__bf16)o[ct][e];
            }
    }
}

// ----------------------------- residual adds (window unpartition for #1)
__global__ __launch_bounds__(256) void residual1_kernel(
        const float* __restrict__ x, const float* __restrict__ mod,
        const float* __restrict__ proj_out, float* __restrict__ x_mid) {
    size_t idx = (size_t)blockIdx.x * 256 + threadIdx.x;
    int row = (int)(idx >> 10), col = (int)(idx & 1023);
    int n = row >> 12, t = row & 4095;
    size_t wr = (size_t)win_row(n, t);
    float g = mod[(size_t)n * 6 * D_MODEL + 2 * D_MODEL + col];
    x_mid[idx] = x[idx] + g * proj_out[wr * D_MODEL + col];
}

__global__ __launch_bounds__(256) void residual2_kernel(
        const float* __restrict__ x_mid, const float* __restrict__ mod,
        const float* __restrict__ mlp_out, float* __restrict__ out) {
    size_t idx = (size_t)blockIdx.x * 256 + threadIdx.x;
    int row = (int)(idx >> 10), col = (int)(idx & 1023);
    int n = row >> 12;
    float g = mod[(size_t)n * 6 * D_MODEL + 5 * D_MODEL + col];
    out[idx] = x_mid[idx] + g * mlp_out[idx];
}

// ---------------------------------------------------------------------------
extern "C" void kernel_launch(void* const* d_in, const int* in_sizes, int n_in,
                              void* d_out, int out_size, void* d_ws, size_t ws_size,
                              hipStream_t stream) {
    const float* x      = (const float*)d_in[0];
    const float* c      = (const float*)d_in[1];
    const float* qkv_w  = (const float*)d_in[2];
    const float* qkv_b  = (const float*)d_in[3];
    const float* proj_w = (const float*)d_in[4];
    const float* proj_b = (const float*)d_in[5];
    const float* rph    = (const float*)d_in[6];
    const float* rpw    = (const float*)d_in[7];
    const float* ada_w  = (const float*)d_in[8];
    const float* ada_b  = (const float*)d_in[9];
    const float* fc1_w  = (const float*)d_in[10];
    const float* fc1_b  = (const float*)d_in[11];
    const float* fc2_w  = (const float*)d_in[12];
    const float* fc2_b  = (const float*)d_in[13];
    float* out = (float*)d_out;

    char* w = (char*)d_ws;
    const size_t o_wqkv = 0;                      // 3072*1024 bf16
    const size_t o_wproj = 6291456;               // 1024*1024 bf16
    const size_t o_wfc1 = 8388608;                // 4096*1024 bf16
    const size_t o_wfc2 = 16777216;               // 1024*4096 bf16
    const size_t o_mod  = 25165824;               // 4*6144 f32
    const size_t o_ain  = 25264128;               // 16384*1024 bf16
    const size_t o_qkv  = 58818560;               // 16384*3072 f32 (reused by h/mlp)
    const size_t o_h    = o_qkv;                  // 16384*4096 bf16
    const size_t o_mlp  = o_qkv + 134217728;      // 16384*1024 f32
    const size_t o_attn = 260145152;              // 16384*1024 bf16
    const size_t o_proj = 293699584;              // 16384*1024 f32
    const size_t o_min  = o_proj;                 // reuse: 16384*1024 bf16
    const size_t o_xmid = 360808448;              // 16384*1024 f32

    __bf16* wqkv  = (__bf16*)(w + o_wqkv);
    __bf16* wproj = (__bf16*)(w + o_wproj);
    __bf16* wfc1  = (__bf16*)(w + o_wfc1);
    __bf16* wfc2  = (__bf16*)(w + o_wfc2);
    float*  modv  = (float*)(w + o_mod);
    __bf16* a_in  = (__bf16*)(w + o_ain);
    float*  qkvf  = (float*)(w + o_qkv);
    __bf16* attn  = (__bf16*)(w + o_attn);
    float*  projo = (float*)(w + o_proj);
    float*  x_mid = (float*)(w + o_xmid);
    __bf16* m_in  = (__bf16*)(w + o_min);
    __bf16* hbuf  = (__bf16*)(w + o_h);
    float*  mlpo  = (float*)(w + o_mlp);

    // 1. weight conversions to bf16
    f2bf_kernel<<<12288, 256, 0, stream>>>(qkv_w, wqkv, 3145728);
    f2bf_kernel<<<4096, 256, 0, stream>>>(proj_w, wproj, 1048576);
    f2bf_kernel<<<16384, 256, 0, stream>>>(fc1_w, wfc1, 4194304);
    f2bf_kernel<<<16384, 256, 0, stream>>>(fc2_w, wfc2, 4194304);
    // 2. adaLN modulation
    mod_kernel<<<dim3(24, 4), 256, 0, stream>>>(c, ada_w, ada_b, modv);
    // 3. LN + modulate (MSA), window-partitioned bf16
    ln_mod_kernel<<<M_ROWS, 256, 0, stream>>>(x, modv, 0, D_MODEL, a_in, 1);
    // 4. QKV GEMM
    gemm_bf16_kernel<0, true, true, false><<<dim3(24, 128), 256, 0, stream>>>(
        a_in, wqkv, qkv_b, qkvf, nullptr, M_ROWS, 3 * D_MODEL, D_MODEL);
    // 5. fused window attention
    attn_kernel<<<NWIN * NH, 256, 0, stream>>>(qkvf, rph, rpw, attn, 0.125f);
    // 6. proj GEMM
    gemm_bf16_kernel<0, true, true, false><<<dim3(8, 128), 256, 0, stream>>>(
        attn, wproj, proj_b, projo, nullptr, M_ROWS, D_MODEL, D_MODEL);
    // 7. gated residual (window unpartition)
    residual1_kernel<<<65536, 256, 0, stream>>>(x, modv, projo, x_mid);
    // 8. LN + modulate (MLP)
    ln_mod_kernel<<<M_ROWS, 256, 0, stream>>>(x_mid, modv, 3 * D_MODEL, 4 * D_MODEL, m_in, 0);
    // 9. fc1 GEMM + GELU -> bf16
    gemm_bf16_kernel<1, true, false, true><<<dim3(32, 128), 256, 0, stream>>>(
        m_in, wfc1, fc1_b, nullptr, hbuf, M_ROWS, 4 * D_MODEL, D_MODEL);
    // 10. fc2 GEMM
    gemm_bf16_kernel<0, true, true, false><<<dim3(8, 128), 256, 0, stream>>>(
        hbuf, wfc2, fc2_b, mlpo, nullptr, M_ROWS, D_MODEL, 4 * D_MODEL);
    // 11. final gated residual
    residual2_kernel<<<65536, 256, 0, stream>>>(x_mid, modv, mlpo, out);
}
